// ArrangeScoreModelGNN_40467181863475
// MI455X (gfx1250) — matrix operand order repaired and checked by reading, not compiled, not run on hardware
//
#include <hip/hip_runtime.h>
#include <stdint.h>

#define N_NODES 50000
#define N_EDGES 800000

typedef __attribute__((ext_vector_type(16))) __bf16 v16bf;
typedef __attribute__((ext_vector_type(8)))  float  v8f;
typedef __attribute__((ext_vector_type(2)))  __bf16 bf16x2;

union Frag { uint32_t u[8]; v16bf v; };
union U32B { uint32_t u; bf16x2 h; };

__device__ __forceinline__ v8f v8f_zero() {
    v8f z = {0.f,0.f,0.f,0.f,0.f,0.f,0.f,0.f};
    return z;
}
// General float atomic max via sign-aware int/uint ordering (buffer init'd to -inf bits)
__device__ __forceinline__ void atomicMaxF32(float* addr, float val) {
    if (!(val < 0.0f)) atomicMax((int*)addr, __float_as_int(val));
    else               atomicMin((unsigned int*)addr, __float_as_uint(val));
}
// Non-negative-only float atomic max (monotonic as int); buffer init'd to 0.0f
__device__ __forceinline__ void atomicMaxF32NonNeg(float* addr, float val) {
    atomicMax((int*)addr, __float_as_int(val));
}

// ---------------------------------------------------------------------------
// Repack fp32 row-major weights [K][Nw] into WMMA bf16 B-fragment layout,
// LANE-MAJOR so each lane's 8 fragment dwords are contiguous (2x b128 loads):
//   out[((kt*NT + nt)*32 + lane)*8 + r]
// lane: col=lane&15, khalf=lane>>4; dword r packs rows k = kt*32+khalf*8+(r>>2)*16+2(r&3)
// (lo) and k+1 (hi), column nt*16+col. Columns >= Nw zero-padded (w22: 4->16).
// foldK > 0 folds the EdgeConv concat algebra into the weights:
//   rows k < foldK become (w[k] - w[k+foldK]) so that
//   [xi, xj-xi] @ W  ==  xi @ (Wtop - Wbot) + xj @ Wbot.
// ---------------------------------------------------------------------------
__global__ void pack_weights(const float* __restrict__ w, uint32_t* __restrict__ out,
                             int Nw, int NT, int KT, int foldK) {
    int idx = blockIdx.x * blockDim.x + threadIdx.x;
    int total = KT * NT * 256;
    if (idx >= total) return;
    int r     = idx & 7;
    int lane  = (idx >> 3) & 31;
    int rem   = idx >> 8;
    int nt    = rem % NT;
    int kt    = rem / NT;
    int col   = lane & 15;
    int khalf = lane >> 4;
    int n     = nt * 16 + col;
    int ka    = kt * 32 + khalf * 8 + ((r >> 2) << 4) + ((r & 3) << 1);
    U32B d;
    d.h = (bf16x2){(__bf16)0.f, (__bf16)0.f};
    if (n < Nw) {
        float lo = w[(size_t)ka * Nw + n];
        float hi = w[(size_t)(ka + 1) * Nw + n];
        if (foldK > 0 && ka < foldK) {
            lo -= w[(size_t)(ka + foldK) * Nw + n];
            hi -= w[(size_t)(ka + 1 + foldK) * Nw + n];
        }
        d.h[0] = (__bf16)lo;
        d.h[1] = (__bf16)hi;
    }
    out[idx] = d.u;
}

// ---------------------------------------------------------------------------
// Node-level features: h0 (init encoder) + cond = [class_feat, sigma_feat],
// written as bf16 into hcat[N][256] = [h0(128) | class(64) | sigma(64)].
// One block (128 threads) per node.
// ---------------------------------------------------------------------------
__global__ __launch_bounds__(128)
void node_kernel(const float* __restrict__ x, const float* __restrict__ geo,
                 const float* __restrict__ t, const int* __restrict__ cat,
                 const float* __restrict__ wg, const float* __restrict__ w_sig,
                 const float* __restrict__ b_sig, const float* __restrict__ embcat,
                 const float* __restrict__ w_i1, const float* __restrict__ b_i1,
                 const float* __restrict__ w_i2, const float* __restrict__ b_i2,
                 __bf16* __restrict__ hcat) {
    int n = blockIdx.x;
    int j = threadIdx.x;
    __shared__ float sh_in[8];
    __shared__ float sh_gf[64];
    __shared__ float sh_t1[128];

    if (j < 4) sh_in[j] = x[(size_t)n * 4 + j];
    if (j >= 4 && j < 6) sh_in[j] = geo[(size_t)n * 2 + (j - 4)];
    if (j < 32) {
        float tp = t[n] * wg[j] * 6.28318530717958647692f;
        sh_gf[j]      = sinf(tp);
        sh_gf[j + 32] = cosf(tp);
    }
    __syncthreads();

    float a = b_i1[j];
    for (int k = 0; k < 6; ++k) a += sh_in[k] * w_i1[k * 128 + j];
    sh_t1[j] = fmaxf(a, 0.f);
    __syncthreads();

    float h = b_i2[j];
    for (int k = 0; k < 128; ++k) h += sh_t1[k] * w_i2[k * 128 + j];
    hcat[(size_t)n * 256 + j] = (__bf16)fmaxf(h, 0.f);

    if (j < 64) {
        int c = cat[n];
        hcat[(size_t)n * 256 + 128 + j] = (__bf16)fmaxf(embcat[c * 64 + j], 0.f);
        float s = b_sig[j];
        for (int k = 0; k < 64; ++k) s += sh_gf[k] * w_sig[k * 64 + j];
        hcat[(size_t)n * 256 + 192 + j] = (__bf16)fmaxf(s, 0.f);
    }
}

__global__ void fill_u32(uint32_t* __restrict__ p, uint32_t v, int n) {
    int i = blockIdx.x * blockDim.x + threadIdx.x;
    if (i < n) p[i] = v;
}

// ---------------------------------------------------------------------------
// EdgeConv with folded weights: layer1 = xi @ W1'[0:256] + xj @ W1'[256:512],
// so A fragments are pure contiguous b128 loads (kt<8 from xi, kt>=8 from xj).
// Per k-step all 8 B fragments are preloaded into bs[8] so the 8-WMMA chain
// can overlap with in-flight loads via partial s_wait_loadcnt.
// layer1 -> bias/ReLU -> LDS -> layer2 -> atomic segment-max scatter onto dst.
// RELU_AGG: conv1 clamps >=0 and uses single-op int atomicMax (agg zero-init).
// 8 waves/block, one 16-edge tile per wave, grid = E/16/8.
// ---------------------------------------------------------------------------
template<int NT2, int AGG_STRIDE, bool RELU_AGG>
__global__ __launch_bounds__(256)
void edgeconv_kernel(const uint16_t* __restrict__ hcat,
                     const int* __restrict__ src, const int* __restrict__ dst,
                     const uint32_t* __restrict__ W1p, const float* __restrict__ b1,
                     const uint32_t* __restrict__ W2p, const float* __restrict__ b2,
                     float* __restrict__ agg) {
    __shared__ __bf16 lds_m2[8][16 * 128];   // 32 KB: layer1 output, bf16 row-major
    __shared__ int    lds_dst[8][16];

    const int wave  = threadIdx.x >> 5;
    const int lane  = threadIdx.x & 31;
    const int ln15  = lane & 15;
    const int khalf = lane >> 4;
    const int tile  = blockIdx.x * 8 + wave;   // grid sized exactly: tile < E/16

    const int e  = tile * 16 + ln15;
    const int es = src[e];
    const int ed = dst[e];
    if (khalf == 0) lds_dst[wave][ln15] = ed;

    const uint16_t* pxi = hcat + (size_t)ed * 256;   // x_i = h[dst]
    const uint16_t* pxj = hcat + (size_t)es * 256;   // x_j = h[src]

    v8f acc[8];
#pragma unroll
    for (int nt = 0; nt < 8; ++nt) acc[nt] = v8f_zero();

    // ---- layer 1: [16,512] x [512,128], K in 16 steps of 32 ----
    for (int kt = 0; kt < 16; ++kt) {
        // folded weights: first 256 K-rows multiply xi, last 256 multiply xj
        const uint16_t* pa = (kt < 8) ? (pxi + kt * 32) : (pxj + (kt - 8) * 32);
        Frag a;
#pragma unroll
        for (int r = 0; r < 8; ++r) {
            const int off = khalf * 8 + ((r >> 2) << 4) + ((r & 3) << 1); // ISA A layout
            a.u[r] = *(const uint32_t*)(pa + off);                        // 2x b128
        }
        Frag bs[8];
#pragma unroll
        for (int nt = 0; nt < 8; ++nt) {
            const uint32_t* wp = W1p + ((size_t)((kt * 8 + nt) * 32 + lane) << 3);
#pragma unroll
            for (int r = 0; r < 8; ++r) bs[nt].u[r] = wp[r];  // 16x b128 clause
        }
#pragma unroll
        for (int nt = 0; nt < 8; ++nt) {
            acc[nt] = __builtin_amdgcn_wmma_f32_16x16x32_bf16(
                false, a.v, false, bs[nt].v, (short)0, acc[nt], false, false);
        }
    }

    // ---- bias + ReLU, spill to LDS in row-major bf16 for layer2 A frags ----
#pragma unroll
    for (int nt = 0; nt < 8; ++nt) {
        const float bias = b1[nt * 16 + ln15];                 // C layout: col = ln15
#pragma unroll
        for (int r = 0; r < 8; ++r) {
            const int m = r + (khalf << 3);                    // C layout: row
            lds_m2[wave][m * 128 + nt * 16 + ln15] = (__bf16)fmaxf(acc[nt][r] + bias, 0.f);
        }
    }
    __syncthreads();

    // ---- layer 2: [16,128] x [128, NT2*16], K in 4 steps of 32 ----
    v8f acc2[NT2];
#pragma unroll
    for (int nt = 0; nt < NT2; ++nt) acc2[nt] = v8f_zero();
#pragma unroll
    for (int kt = 0; kt < 4; ++kt) {
        Frag a;
        const int kbase = kt * 32 + khalf * 8;
#pragma unroll
        for (int r = 0; r < 8; ++r) {
            const int f = kbase + ((r >> 2) << 4) + ((r & 3) << 1);
            a.u[r] = *(const uint32_t*)&lds_m2[wave][ln15 * 128 + f];
        }
        Frag bs[NT2];
#pragma unroll
        for (int nt = 0; nt < NT2; ++nt) {
            const uint32_t* wp = W2p + ((size_t)((kt * NT2 + nt) * 32 + lane) << 3);
#pragma unroll
            for (int r = 0; r < 8; ++r) bs[nt].u[r] = wp[r];
        }
#pragma unroll
        for (int nt = 0; nt < NT2; ++nt) {
            acc2[nt] = __builtin_amdgcn_wmma_f32_16x16x32_bf16(
                false, a.v, false, bs[nt].v, (short)0, acc2[nt], false, false);
        }
    }

    // ---- bias + atomic segment-max scatter onto dst nodes ----
#pragma unroll
    for (int nt = 0; nt < NT2; ++nt) {
        const int  col   = nt * 16 + ln15;
        const bool valid = (AGG_STRIDE >= 128) || (ln15 < 4);  // conv2: only 4 real cols
        const float bias = valid ? b2[col] : 0.f;
#pragma unroll
        for (int r = 0; r < 8; ++r) {
            if (valid) {
                const int m    = r + (khalf << 3);
                const int node = lds_dst[wave][m];
                float v = acc2[nt][r] + bias;
                float* p = agg + (size_t)node * AGG_STRIDE + col;
                if (RELU_AGG) atomicMaxF32NonNeg(p, fmaxf(v, 0.f)); // conv1: relu'd later
                else          atomicMaxF32(p, v);                   // conv2: signed max
            }
        }
    }
}

// conv1 post: h[:,0:128] <- relu(segmax) as bf16 (agg1 zero-init, already >= 0)
__global__ void fixup1(const float* __restrict__ agg1, __bf16* __restrict__ hcat, int total) {
    int i = blockIdx.x * blockDim.x + threadIdx.x;
    if (i >= total) return;
    int n = i >> 7, j = i & 127;
    hcat[(size_t)n * 256 + j] = (__bf16)agg1[i];
}

// conv2 post: out = where(-inf,0,agg2) / (std(t)+1e-7)
__global__ void fixup2(const float* __restrict__ agg2, const float* __restrict__ t,
                       float* __restrict__ out, int total) {
    int i = blockIdx.x * blockDim.x + threadIdx.x;
    if (i >= total) return;
    int n = i >> 2;
    float v = agg2[i];
    if (__float_as_uint(v) == 0xFF800000u) v = 0.f;
    const float ln_sig = 3.21887582486820075f;               // log(25)
    float tt = t[n];
    float stddev = sqrtf((expf(2.f * tt * ln_sig) - 1.f) / (2.f * ln_sig));
    out[i] = v / (stddev + 1e-7f);
}

extern "C" void kernel_launch(void* const* d_in, const int* in_sizes, int n_in,
                              void* d_out, int out_size, void* d_ws, size_t ws_size,
                              hipStream_t stream) {
    (void)in_sizes; (void)n_in; (void)out_size; (void)ws_size;
    const float* x      = (const float*)d_in[0];
    const float* geo    = (const float*)d_in[1];
    const float* t      = (const float*)d_in[2];
    const int*   cat    = (const int*)  d_in[3];
    const int*   ei     = (const int*)  d_in[4];
    const float* wg     = (const float*)d_in[5];
    const float* w_sig  = (const float*)d_in[6];
    const float* b_sig  = (const float*)d_in[7];
    const float* embcat = (const float*)d_in[8];
    const float* w_i1   = (const float*)d_in[9];
    const float* b_i1   = (const float*)d_in[10];
    const float* w_i2   = (const float*)d_in[11];
    const float* b_i2   = (const float*)d_in[12];
    const float* w11    = (const float*)d_in[13];
    const float* b11    = (const float*)d_in[14];
    const float* w12    = (const float*)d_in[15];
    const float* b12    = (const float*)d_in[16];
    const float* w21    = (const float*)d_in[17];
    const float* b21    = (const float*)d_in[18];
    const float* w22    = (const float*)d_in[19];
    const float* b22    = (const float*)d_in[20];

    // workspace layout (bytes)
    uint8_t* ws = (uint8_t*)d_ws;
    uint32_t* W11p = (uint32_t*)(ws + 0);        // 16*8*256*4  = 131072
    uint32_t* W12p = (uint32_t*)(ws + 131072);   //  4*8*256*4  =  32768
    uint32_t* W21p = (uint32_t*)(ws + 163840);   // 131072
    uint32_t* W22p = (uint32_t*)(ws + 294912);   //  4*1*256*4  =   4096
    uint16_t* hcat = (uint16_t*)(ws + 299008);   // N*256*2     = 25600000
    float*    agg1 = (float*)   (ws + 299008 + 25600000);  // N*128*4 = 25600000
    float*    agg2 = (float*)   (ws + 299008 + 51200000);  // N*4*4   =   800000

    const int* srcp = ei;
    const int* dstp = ei + N_EDGES;

    pack_weights<<<(16*8*256 + 255)/256, 256, 0, stream>>>(w11, W11p, 128, 8, 16, 256);
    pack_weights<<<( 4*8*256 + 255)/256, 256, 0, stream>>>(w12, W12p, 128, 8,  4,   0);
    pack_weights<<<(16*8*256 + 255)/256, 256, 0, stream>>>(w21, W21p, 128, 8, 16, 256);
    pack_weights<<<( 4*1*256 + 255)/256, 256, 0, stream>>>(w22, W22p,   4, 1,  4,   0);

    node_kernel<<<N_NODES, 128, 0, stream>>>(x, geo, t, cat, wg, w_sig, b_sig,
                                             embcat, w_i1, b_i1, w_i2, b_i2,
                                             (__bf16*)hcat);

    fill_u32<<<(N_NODES*128 + 255)/256, 256, 0, stream>>>((uint32_t*)agg1, 0u, N_NODES*128);
    fill_u32<<<(N_NODES*4   + 255)/256, 256, 0, stream>>>((uint32_t*)agg2, 0xFF800000u, N_NODES*4);

    // 50000 tiles of 16 edges, 8 waves/block
    edgeconv_kernel<8, 128, true><<<N_EDGES/16/8, 256, 0, stream>>>(
        hcat, srcp, dstp, W11p, b11, W12p, b12, agg1);
    fixup1<<<(N_NODES*128 + 255)/256, 256, 0, stream>>>(agg1, (__bf16*)hcat, N_NODES*128);
    edgeconv_kernel<1, 4, false><<<N_EDGES/16/8, 256, 0, stream>>>(
        hcat, srcp, dstp, W21p, b21, W22p, b22, agg2);
    fixup2<<<(N_NODES*4 + 255)/256, 256, 0, stream>>>(agg2, t, (float*)d_out, N_NODES*4);
}
